// SparseMoEBlock_67765993997269
// MI455X (gfx1250) — matrix operand order repaired and checked by reading, not compiled, not run on hardware
//
#include <hip/hip_runtime.h>
#include <hip/hip_bf16.h>
#include <math.h>

// ---------------- problem constants (match reference) ----------------
constexpr int Bc = 4;     // batch
constexpr int Sc = 4096;  // sequence
constexpr int Dc = 1024;  // model dim
constexpr int Ec = 8;     // experts
constexpr int Ic = 2048;  // mlp inner dim
constexpr int Kc = 1024;  // tokens per expert = S/E*CAPACITY

typedef __attribute__((ext_vector_type(16))) __bf16 v16bf;
typedef __attribute__((ext_vector_type(8)))  float  v8f;
typedef __attribute__((ext_vector_type(4)))  __bf16 v4bf;

union FragU { v16bf bf; uint4 q[2]; };

// LDS tiles are 128 rows x 32 bf16 cols, padded to 40 elems/row (80B) to
// spread bank usage (20 dwords/row -> 16 distinct row-start banks).
constexpr int LDT = 40;
constexpr int TILE_ELEMS = 128 * LDT;

// A-fragment (16x32 bf16): lanes 0-15 = rows, hold K 0-7 & 16-23;
// lanes 16-31 hold K 8-15 & 24-31 (ISA 7.12.2).
__device__ __forceinline__ v16bf load_frag_A(const __bf16* tile, int row0, int lane) {
  const int m = lane & 15;
  const int h = lane >> 4;
  const char* row = (const char*)(tile + (size_t)(row0 + m) * LDT);
  FragU f;
  f.q[0] = *(const uint4*)(row + h * 16);
  f.q[1] = *(const uint4*)(row + 32 + h * 16);
  return f.bf;
}

// B-fragment (32x16 bf16), B tile staged transposed in LDS as [n][k]:
// lanes 0-15 = cols n, hold K 0-7 & 16-23; lanes 16-31 hold K 8-15 & 24-31.
__device__ __forceinline__ v16bf load_frag_B(const __bf16* tile, int row0, int lane) {
  const int n = lane & 15;
  const int h = lane >> 4;
  const char* row = (const char*)(tile + (size_t)(row0 + n) * LDT);
  FragU f;
  f.q[0] = *(const uint4*)(row + h * 16);
  f.q[1] = *(const uint4*)(row + 32 + h * 16);
  return f.bf;
}

// Cooperative 128x32 bf16 tile staging, split into global-fetch (to regs)
// and LDS-commit so the fetch for tile t+1 overlaps the WMMAs of tile t.
struct TileRegs { uint4 a, b; };

__device__ __forceinline__ TileRegs fetch_tile(const __bf16* src, size_t ld, int tid) {
  const int r = tid >> 1;
  const int c = (tid & 1) << 4;
  const uint4* g = (const uint4*)(src + (size_t)r * ld + c);
  TileRegs t;
  t.a = g[0];
  t.b = g[1];
  return t;
}

__device__ __forceinline__ void commit_tile(__bf16* dst, TileRegs t, int tid) {
  const int r = tid >> 1;
  const int c = (tid & 1) << 4;
  *(uint4*)(dst + r * LDT + c)     = t.a;
  *(uint4*)(dst + r * LDT + c + 8) = t.b;
}

// ---------------- 1) router: logits -> softmax over experts ----------------
// one wave32 per token; affT layout [B,E,S]
__global__ __launch_bounds__(256) void router_kernel(const float* __restrict__ x,
                                                     const float* __restrict__ gw,
                                                     float* __restrict__ affT) {
  const int gwave = (int)((blockIdx.x * blockDim.x + threadIdx.x) >> 5);
  const int lane  = threadIdx.x & 31;
  const int b = gwave / Sc;
  const int s = gwave % Sc;
  const float* xr = x + ((size_t)b * Sc + s) * Dc;

  float acc[Ec];
#pragma unroll
  for (int e = 0; e < Ec; ++e) acc[e] = 0.0f;

  for (int d = lane; d < Dc; d += 32) {
    const float xv = xr[d];
    const float* g = gw + (size_t)d * Ec;
#pragma unroll
    for (int e = 0; e < Ec; ++e) acc[e] += xv * g[e];
  }
#pragma unroll
  for (int off = 16; off >= 1; off >>= 1)
#pragma unroll
    for (int e = 0; e < Ec; ++e) acc[e] += __shfl_xor(acc[e], off, 32);

  float mx = acc[0];
#pragma unroll
  for (int e = 1; e < Ec; ++e) mx = fmaxf(mx, acc[e]);
  float p[Ec], den = 0.0f;
#pragma unroll
  for (int e = 0; e < Ec; ++e) { p[e] = __expf(acc[e] - mx); den += p[e]; }
  const float inv = 1.0f / den;
  if (lane < Ec)
    affT[((size_t)b * Ec + lane) * Sc + s] = p[lane] * inv;
}

// ---------------- 2) exact-rank top-K per (b,e) ----------------
// rank(i) = #{j: v_j > v_i} + #{j<i: v_j == v_i}  (stable == jax top_k order)
__global__ __launch_bounds__(256) void topk_kernel(const float* __restrict__ affT,
                                                   int* __restrict__ idxb,
                                                   float* __restrict__ gatb) {
  __shared__ float sv[Sc];  // 16 KB
  const int be = blockIdx.x;
  const float* a = affT + (size_t)be * Sc;
  for (int i = threadIdx.x; i < Sc; i += 256) sv[i] = a[i];
  __syncthreads();
  for (int i = threadIdx.x; i < Sc; i += 256) {
    const float v = sv[i];
    int rank = 0;
    for (int j = 0; j < Sc; ++j) {
      const float vj = sv[j];
      rank += (vj > v) || (vj == v && j < i);
    }
    if (rank < Kc) {
      idxb[(size_t)be * Kc + rank] = i;
      gatb[(size_t)be * Kc + rank] = v;
    }
  }
}

// ---------------- 3) transpose + fp32->bf16 weight convert ----------------
// in: [nmat, R, C] fp32 ; out: [nmat, C, R] bf16  (so GEMM B reads are K-contig)
__global__ void transcvt_kernel(const float* __restrict__ in, __bf16* __restrict__ out,
                                int R, int C) {
  __shared__ float tile[32][33];
  const int mat = blockIdx.z;
  const int r0 = blockIdx.y * 32, c0 = blockIdx.x * 32;
  const float* src = in + (size_t)mat * R * C;
  __bf16* dst = out + (size_t)mat * R * C;
  const int tx = threadIdx.x, ty = threadIdx.y;  // 32 x 8
#pragma unroll
  for (int i = 0; i < 32; i += 8)
    tile[ty + i][tx] = src[(size_t)(r0 + ty + i) * C + (c0 + tx)];
  __syncthreads();
#pragma unroll
  for (int i = 0; i < 32; i += 8)
    dst[(size_t)(c0 + ty + i) * R + (r0 + tx)] = (__bf16)tile[tx][ty + i];
}

// ---------------- 4) gather selected tokens -> bf16 Xsel ----------------
__global__ void gather_kernel(const float* __restrict__ x, const int* __restrict__ idxb,
                              __bf16* __restrict__ Xsel) {
  const size_t t = (size_t)blockIdx.x * blockDim.x + threadIdx.x;  // over B*E*K*(D/4)
  const int d4 = (int)(t & (Dc / 4 - 1));
  const size_t row = t >> 8;           // /(D/4)=256 ; row = be*K + k
  const int be = (int)(row >> 10);     // /K=1024
  const int b = be >> 3;               // /E
  const int tok = idxb[row];
  const float4 v = *(const float4*)(x + ((size_t)b * Sc + tok) * Dc + (size_t)d4 * 4);
  v4bf o;
  o[0] = (__bf16)v.x; o[1] = (__bf16)v.y; o[2] = (__bf16)v.z; o[3] = (__bf16)v.w;
  *(v4bf*)(Xsel + t * 4) = o;
}

// ---------------- 5) H = silu(Xsel*Wg) .* (Xsel*Wu)  [WMMA bf16] ----------------
// grid (I/128, K/128, B*E), 256 threads = 8 waves, wave tile 64x32 (4x2 WMMA
// C-tiles per GEMM), double-buffered LDS (2 x 3 x 10KB = 60KB).
__global__ __launch_bounds__(256) void gemm_h_kernel(const __bf16* __restrict__ Xsel,
                                                     const __bf16* __restrict__ WgT,
                                                     const __bf16* __restrict__ WuT,
                                                     __bf16* __restrict__ H) {
  __shared__ alignas(16) __bf16 sA[2][TILE_ELEMS];
  __shared__ alignas(16) __bf16 sBg[2][TILE_ELEMS];
  __shared__ alignas(16) __bf16 sBu[2][TILE_ELEMS];

  const int be = blockIdx.z;
  const int e  = be & (Ec - 1);
  const int m0 = blockIdx.y * 128;
  const int n0 = blockIdx.x * 128;
  const int tid  = threadIdx.x;
  const int lane = tid & 31;
  const int wave = tid >> 5;
  const int wm = (wave >> 2) * 64;
  const int wn = (wave & 3) * 32;

  const __bf16* Ab = Xsel + ((size_t)be * Kc + m0) * Dc;
  const __bf16* Bg = WgT + ((size_t)e * Ic + n0) * Dc;
  const __bf16* Bu = WuT + ((size_t)e * Ic + n0) * Dc;

  v8f cg[4][2] = {};
  v8f cu[4][2] = {};

  // prologue: stage tile 0
  {
    TileRegs ra = fetch_tile(Ab, Dc, tid);
    TileRegs rg = fetch_tile(Bg, Dc, tid);
    TileRegs ru = fetch_tile(Bu, Dc, tid);
    commit_tile(sA[0], ra, tid);
    commit_tile(sBg[0], rg, tid);
    commit_tile(sBu[0], ru, tid);
  }
  __syncthreads();

  for (int kk = 0; kk < Dc; kk += 32) {
    const int cur = (kk >> 5) & 1;
    const int nxt = cur ^ 1;
    const bool more = (kk + 32) < Dc;

    // prefetch next tile global->regs while this tile computes
    TileRegs ra, rg, ru;
    if (more) {
      ra = fetch_tile(Ab + kk + 32, Dc, tid);
      rg = fetch_tile(Bg + kk + 32, Dc, tid);
      ru = fetch_tile(Bu + kk + 32, Dc, tid);
    }

    v16bf a[4], bg[2], bu[2];
#pragma unroll
    for (int t = 0; t < 4; ++t) a[t] = load_frag_A(sA[cur], wm + t * 16, lane);
#pragma unroll
    for (int s = 0; s < 2; ++s) {
      bg[s] = load_frag_B(sBg[cur], wn + s * 16, lane);
      bu[s] = load_frag_B(sBu[cur], wn + s * 16, lane);
    }
#pragma unroll
    for (int t = 0; t < 4; ++t)
#pragma unroll
      for (int s = 0; s < 2; ++s) {
        cg[t][s] = __builtin_amdgcn_wmma_f32_16x16x32_bf16(
            false, a[t], false, bg[s], (short)0, cg[t][s], false, false);
        cu[t][s] = __builtin_amdgcn_wmma_f32_16x16x32_bf16(
            false, a[t], false, bu[s], (short)0, cu[t][s], false, false);
      }

    if (more) {
      commit_tile(sA[nxt], ra, tid);
      commit_tile(sBg[nxt], rg, tid);
      commit_tile(sBu[nxt], ru, tid);
    }
    __syncthreads();
  }

  const int mh = (lane >> 4) << 3;  // lanes 16-31 hold rows +8
  const int nl = lane & 15;
#pragma unroll
  for (int t = 0; t < 4; ++t)
#pragma unroll
    for (int s = 0; s < 2; ++s)
#pragma unroll
      for (int r = 0; r < 8; ++r) {
        const float g = cg[t][s][r];
        const float u = cu[t][s][r];
        // silu(g)*u with fast v_rcp_f32 (bf16 output makes 1-ulp rcp harmless)
        const float sig = __builtin_amdgcn_rcpf(1.0f + __expf(-g));
        const float h = g * sig * u;
        const int m = m0 + wm + t * 16 + mh + r;
        const int n = n0 + wn + s * 16 + nl;
        H[((size_t)be * Kc + m) * Ic + n] = (__bf16)h;
      }
}

// ---------------- 6) Out = H*Wd, scale by gating, atomic scatter-add ----------------
// grid (D/128, K/128, B*E), double-buffered LDS (2 x 2 x 10KB = 40KB)
__global__ __launch_bounds__(256) void gemm_out_kernel(const __bf16* __restrict__ H,
                                                       const __bf16* __restrict__ WdT,
                                                       const int* __restrict__ idxb,
                                                       const float* __restrict__ gatb,
                                                       float* __restrict__ out) {
  __shared__ alignas(16) __bf16 sA[2][TILE_ELEMS];
  __shared__ alignas(16) __bf16 sB[2][TILE_ELEMS];

  const int be = blockIdx.z;
  const int b  = be >> 3;
  const int e  = be & (Ec - 1);
  const int m0 = blockIdx.y * 128;
  const int n0 = blockIdx.x * 128;
  const int tid  = threadIdx.x;
  const int lane = tid & 31;
  const int wave = tid >> 5;
  const int wm = (wave >> 2) * 64;
  const int wn = (wave & 3) * 32;

  const __bf16* Ab = H   + ((size_t)be * Kc + m0) * Ic;
  const __bf16* Bb = WdT + ((size_t)e * Dc + n0) * Ic;

  v8f c[4][2] = {};

  {
    TileRegs ra = fetch_tile(Ab, Ic, tid);
    TileRegs rb = fetch_tile(Bb, Ic, tid);
    commit_tile(sA[0], ra, tid);
    commit_tile(sB[0], rb, tid);
  }
  __syncthreads();

  for (int kk = 0; kk < Ic; kk += 32) {
    const int cur = (kk >> 5) & 1;
    const int nxt = cur ^ 1;
    const bool more = (kk + 32) < Ic;

    TileRegs ra, rb;
    if (more) {
      ra = fetch_tile(Ab + kk + 32, Ic, tid);
      rb = fetch_tile(Bb + kk + 32, Ic, tid);
    }

    v16bf a[4], bw[2];
#pragma unroll
    for (int t = 0; t < 4; ++t) a[t] = load_frag_A(sA[cur], wm + t * 16, lane);
#pragma unroll
    for (int s = 0; s < 2; ++s) bw[s] = load_frag_B(sB[cur], wn + s * 16, lane);
#pragma unroll
    for (int t = 0; t < 4; ++t)
#pragma unroll
      for (int s = 0; s < 2; ++s)
        c[t][s] = __builtin_amdgcn_wmma_f32_16x16x32_bf16(
            false, a[t], false, bw[s], (short)0, c[t][s], false, false);

    if (more) {
      commit_tile(sA[nxt], ra, tid);
      commit_tile(sB[nxt], rb, tid);
    }
    __syncthreads();
  }

  const int mh = (lane >> 4) << 3;
  const int nl = lane & 15;
#pragma unroll
  for (int t = 0; t < 4; ++t)
#pragma unroll
    for (int s = 0; s < 2; ++s)
#pragma unroll
      for (int r = 0; r < 8; ++r) {
        const int m = m0 + wm + t * 16 + mh + r;
        const int n = n0 + wn + s * 16 + nl;
        const int   tok = idxb[(size_t)be * Kc + m];
        const float gv  = gatb[(size_t)be * Kc + m];
        atomicAdd(out + ((size_t)b * Sc + tok) * Dc + n, c[t][s][r] * gv);
      }
}

// ---------------- host launcher ----------------
extern "C" void kernel_launch(void* const* d_in, const int* in_sizes, int n_in,
                              void* d_out, int out_size, void* d_ws, size_t ws_size,
                              hipStream_t stream) {
  const float* x  = (const float*)d_in[0];  // [B,S,D]
  const float* gw = (const float*)d_in[1];  // [D,E]
  const float* Wg = (const float*)d_in[2];  // [E,D,I]
  const float* Wu = (const float*)d_in[3];  // [E,D,I]
  const float* Wd = (const float*)d_in[4];  // [E,I,D]
  float* out = (float*)d_out;               // [B,S,D]

  // workspace layout (~289 MB total)
  char* ws = (char*)d_ws;
  size_t off = 0;
  auto take = [&](size_t bytes) {
    void* p = ws + off;
    off += (bytes + 255) & ~(size_t)255;
    return p;
  };
  float*  affT = (float*)take((size_t)Bc * Ec * Sc * 4);       // [B,E,S]
  int*    idxb = (int*)take((size_t)Bc * Ec * Kc * 4);         // [B,E,K]
  float*  gatb = (float*)take((size_t)Bc * Ec * Kc * 4);       // [B,E,K]
  __bf16* WgT  = (__bf16*)take((size_t)Ec * Ic * Dc * 2);      // [E,I,D]
  __bf16* WuT  = (__bf16*)take((size_t)Ec * Ic * Dc * 2);      // [E,I,D]
  __bf16* WdT  = (__bf16*)take((size_t)Ec * Dc * Ic * 2);      // [E,D,I]
  __bf16* Xsel = (__bf16*)take((size_t)Bc * Ec * Kc * Dc * 2); // [BE,K,D]
  __bf16* Hbuf = (__bf16*)take((size_t)Bc * Ec * Kc * Ic * 2); // [BE,K,I]
  (void)ws_size;

  hipMemsetAsync(d_out, 0, (size_t)Bc * Sc * Dc * sizeof(float), stream);

  // router: 8 wave32 tokens per 256-thread block
  router_kernel<<<(Bc * Sc) / 8, 256, 0, stream>>>(x, gw, affT);
  // top-K per (b,e)
  topk_kernel<<<Bc * Ec, 256, 0, stream>>>(affT, idxb, gatb);
  // weight transpose+convert: Wg/Wu [D,I]->[I,D], Wd [I,D]->[D,I]
  transcvt_kernel<<<dim3(Ic / 32, Dc / 32, Ec), dim3(32, 8), 0, stream>>>(Wg, WgT, Dc, Ic);
  transcvt_kernel<<<dim3(Ic / 32, Dc / 32, Ec), dim3(32, 8), 0, stream>>>(Wu, WuT, Dc, Ic);
  transcvt_kernel<<<dim3(Dc / 32, Ic / 32, Ec), dim3(32, 8), 0, stream>>>(Wd, WdT, Ic, Dc);
  // gather tokens (float4 -> 4x bf16 per thread)
  gather_kernel<<<(int)(((size_t)Bc * Ec * Kc * Dc / 4) / 256), 256, 0, stream>>>(x, idxb, Xsel);
  // fused SwiGLU GEMM pair
  gemm_h_kernel<<<dim3(Ic / 128, Kc / 128, Bc * Ec), 256, 0, stream>>>(Xsel, WgT, WuT, Hbuf);
  // down-projection + weighted scatter-add
  gemm_out_kernel<<<dim3(Dc / 128, Kc / 128, Bc * Ec), 256, 0, stream>>>(Hbuf, WdT, idxb, gatb, out);
}
